// SharedTrunkPEER_75926431859380
// MI455X (gfx1250) — compile-verified, matrix operand order checked by reading, not run on hardware
//
#include <hip/hip_runtime.h>
#include <hip/hip_bf16.h>
#include <math.h>

// ---------------------------------------------------------------------------
// SharedTrunkPEER for MI455X (gfx1250, wave32, WMMA).
//   D=1024, H=8, K=16, S=512, HD=128, tokens = B*T = 8192.
// GEMMs: v_wmma_f32_16x16x32_bf16, A-tile staged to LDS via
// global_load_async_to_lds_b128 (ASYNCcnt), double-buffered K-loop.
// ---------------------------------------------------------------------------

#define NTOK   8192
#define DMODEL 1024
#define NHEAD  8
#define HD     128
#define SKEYS  512
#define TOPK   16
#define NEGBIG (-3.0e38f)

#define GEMM_K   1024
#define A_STRIDE 1032   // 1024 + 8 elem pad -> 2064 B row stride = 4-bank skew

typedef __attribute__((ext_vector_type(16))) __bf16 v16bf;
typedef __attribute__((ext_vector_type(8)))  float  v8f;

union FragBF { v16bf v; uint4 u4[2]; };

// Butterfly argmax across the 32 lanes of a wave (value,index), ties -> lower idx.
__device__ __forceinline__ void waveArgmax(float& bv, int& bi) {
#pragma unroll
  for (int off = 16; off > 0; off >>= 1) {
    float ov = __shfl_xor(bv, off, 32);
    int   oi = __shfl_xor(bi, off, 32);
    if (ov > bv || (ov == bv && oi < bi)) { bv = ov; bi = oi; }
  }
}

// ---------------------------------------------------------------------------
// Kernel 1: f32 -> bf16 pack
// ---------------------------------------------------------------------------
__global__ void cvt_f32_bf16(const float* __restrict__ in, __bf16* __restrict__ out, int n) {
  int i = blockIdx.x * blockDim.x + threadIdx.x;
  if (i < n) out[i] = (__bf16)in[i];
}

// ---------------------------------------------------------------------------
// Kernel 2/4: C(8192x1024 f32) = A(8192x1024 bf16) * W(1024x1024 bf16)^T.
// Block = 8 waves = 16M x 512N; wave = 16M x 64N (4 accumulators).
//  - A tile (16x1024) staged to LDS once per block via async-to-LDS (ASYNCcnt)
//  - K-loop double-buffered: WMMA consumes fragments loaded one stage earlier
// ---------------------------------------------------------------------------
__global__ __launch_bounds__(256) void gemm_nt_bf16(
    const __bf16* __restrict__ A, const __bf16* __restrict__ W,
    float* __restrict__ outF, __bf16* __restrict__ outBF)
{
  __shared__ __bf16 sA[16 * A_STRIDE];   // ~32.25 KB of the 320 KB WGP LDS

  const int lane  = threadIdx.x & 31;
  const int wv    = threadIdx.x >> 5;
  const int half  = lane >> 4;        // 0: K {0..7,16..23}, 1: K {8..15,24..31}
  const int l16   = lane & 15;
  const int half8 = half * 8;
  const int m0 = blockIdx.x * 16;
  const int n0 = blockIdx.y * 512 + wv * 64;

  // ---- stage A tile (16 x 1024 bf16 = 32 KB) into LDS asynchronously ----
  // 2048 16-byte chunks, 8 per thread.
#pragma unroll
  for (int i = 0; i < 8; ++i) {
    int idx  = threadIdx.x + i * 256;
    int row  = idx >> 7;          // / 128 chunks per row
    int col8 = idx & 127;         // 16B chunk within row
    const __bf16* g = A + (size_t)(m0 + row) * GEMM_K + col8 * 8;
    unsigned lds = (unsigned)(uintptr_t)(&sA[row * A_STRIDE + col8 * 8]);
    asm volatile("global_load_async_to_lds_b128 %0, %1, off"
                 :: "v"(lds), "v"(g) : "memory");
  }
  asm volatile("s_wait_asynccnt 0x0" ::: "memory");
  __syncthreads();

  const __bf16* aRowL = &sA[l16 * A_STRIDE];

  auto loadA = [&](FragBF& fa, int k0) {
    fa.u4[0] = *reinterpret_cast<const uint4*>(aRowL + k0 + half8);        // ds_load_b128
    fa.u4[1] = *reinterpret_cast<const uint4*>(aRowL + k0 + 16 + half8);
  };
  auto loadB = [&](FragBF (&fb)[4], int k0) {
#pragma unroll
    for (int nt = 0; nt < 4; ++nt) {
      const __bf16* bRow = W + (size_t)(n0 + nt * 16 + l16) * GEMM_K;
      fb[nt].u4[0] = *reinterpret_cast<const uint4*>(bRow + k0 + half8);
      fb[nt].u4[1] = *reinterpret_cast<const uint4*>(bRow + k0 + 16 + half8);
    }
  };

  v8f c[4] = {};
  FragBF fa0, fa1;
  FragBF fb0[4], fb1[4];

  loadA(fa0, 0);
  loadB(fb0, 0);

  for (int k0 = 0; k0 < GEMM_K; k0 += 64) {
    // prefetch stage 1 (k0+32) while stage 0 computes
    loadA(fa1, k0 + 32);
    loadB(fb1, k0 + 32);
#pragma unroll
    for (int nt = 0; nt < 4; ++nt)
      c[nt] = __builtin_amdgcn_wmma_f32_16x16x32_bf16(
          false, fa0.v, false, fb0[nt].v, (short)0, c[nt], false, false);

    if (k0 + 64 < GEMM_K) {
      loadA(fa0, k0 + 64);
      loadB(fb0, k0 + 64);
      __builtin_prefetch(W + (size_t)(n0 + l16) * GEMM_K + k0 + 96, 0, 1);
    }
#pragma unroll
    for (int nt = 0; nt < 4; ++nt)
      c[nt] = __builtin_amdgcn_wmma_f32_16x16x32_bf16(
          false, fa1.v, false, fb1[nt].v, (short)0, c[nt], false, false);
  }

#pragma unroll
  for (int nt = 0; nt < 4; ++nt) {
#pragma unroll
    for (int vv = 0; vv < 8; ++vv) {
      int m = m0 + vv + 8 * half;          // C layout: M = vgpr + 8*(lane/16)
      int n = n0 + nt * 16 + l16;          //           N = lane%16
      float val = c[nt][vv];
      if (outF)  outF[(size_t)m * DMODEL + n]  = val;
      if (outBF) outBF[(size_t)m * DMODEL + n] = (__bf16)val;
    }
  }
}

// ---------------------------------------------------------------------------
// Kernel 3: fused PEER select per (16-token tile, head).
//   WMMA scores_a -> LDS -> top16/token; WMMA scores_b (LDS reuse) -> top16;
//   product-key top16 of 256; softmax; gather expert_v; * sigmoid(u.h);
//   write bf16 merged[token][head*128 + d].
// ---------------------------------------------------------------------------
__global__ __launch_bounds__(256) void peer_select(
    const __bf16* __restrict__ hbf, const float* __restrict__ hf,
    const __bf16* __restrict__ keysA, const __bf16* __restrict__ keysB,
    const float* __restrict__ uvec, const float* __restrict__ expert_v,
    __bf16* __restrict__ merged)
{
  __shared__ float sS[16 * SKEYS];          // 32 KB score tile
  __shared__ float sTopAv[16][TOPK];
  __shared__ int   sTopAi[16][TOPK];
  __shared__ float sTopBv[8][TOPK];
  __shared__ int   sTopBi[8][TOPK];

  const int lane  = threadIdx.x & 31;
  const int wv    = threadIdx.x >> 5;
  const int half  = lane >> 4;
  const int l16   = lane & 15;
  const int half8 = half * 8;
  const int t0    = blockIdx.x * 16;
  const int head  = blockIdx.y;

  // ---- WMMA: sS[16][512] = h_tile (16x128) @ keys(head)^T (128x512) ----
  auto computeScores = [&](const __bf16* keys) {
    v8f c[4] = {};
    const __bf16* aRow = hbf + (size_t)(t0 + l16) * DMODEL + (size_t)head * HD;
    const int n0 = wv * 64;
#pragma unroll
    for (int k0 = 0; k0 < HD; k0 += 32) {
      FragBF fa;
      fa.u4[0] = *reinterpret_cast<const uint4*>(aRow + k0 + half8);
      fa.u4[1] = *reinterpret_cast<const uint4*>(aRow + k0 + 16 + half8);
#pragma unroll
      for (int nt = 0; nt < 4; ++nt) {
        const __bf16* bRow = keys + (size_t)(n0 + nt * 16 + l16) * HD;
        FragBF fb;
        fb.u4[0] = *reinterpret_cast<const uint4*>(bRow + k0 + half8);
        fb.u4[1] = *reinterpret_cast<const uint4*>(bRow + k0 + 16 + half8);
        c[nt] = __builtin_amdgcn_wmma_f32_16x16x32_bf16(
            false, fa.v, false, fb.v, (short)0, c[nt], false, false);
      }
    }
#pragma unroll
    for (int nt = 0; nt < 4; ++nt)
#pragma unroll
      for (int vv = 0; vv < 8; ++vv)
        sS[(vv + 8 * half) * SKEYS + n0 + nt * 16 + l16] = c[nt][vv];
  };

  // ================= phase 1: scores_a =================
  computeScores(keysA + (size_t)head * SKEYS * HD);
  __syncthreads();

  // ================= phase 2: top16 of scores_a (2 tokens per wave) =========
  for (int s = 0; s < 2; ++s) {
    const int tl = 2 * wv + s;
    float lv[16];
#pragma unroll
    for (int j = 0; j < 16; ++j) lv[j] = sS[tl * SKEYS + lane + 32 * j];
#pragma unroll
    for (int k = 0; k < TOPK; ++k) {
      float bv = lv[0]; int bi = lane;
#pragma unroll
      for (int j = 1; j < 16; ++j) {
        float v = lv[j]; int idx = lane + 32 * j;
        if (v > bv) { bv = v; bi = idx; }
      }
      waveArgmax(bv, bi);
      if (lane == 0) { sTopAv[tl][k] = bv; sTopAi[tl][k] = bi; }
#pragma unroll
      for (int j = 0; j < 16; ++j) if (bi == lane + 32 * j) lv[j] = NEGBIG;
    }
  }
  __syncthreads();

  // ================= phase 3: scores_b (reuse LDS) =================
  computeScores(keysB + (size_t)head * SKEYS * HD);
  __syncthreads();

  // ================= phase 4: top16_b, product keys, softmax, gather ========
  for (int s = 0; s < 2; ++s) {
    const int tl = 2 * wv + s;
    const int token = t0 + tl;

    // top16 of scores_b (results in wave-private sTopB rows)
    {
      float lv[16];
#pragma unroll
      for (int j = 0; j < 16; ++j) lv[j] = sS[tl * SKEYS + lane + 32 * j];
#pragma unroll
      for (int k = 0; k < TOPK; ++k) {
        float bv = lv[0]; int bi = lane;
#pragma unroll
        for (int j = 1; j < 16; ++j) {
          float v = lv[j]; int idx = lane + 32 * j;
          if (v > bv) { bv = v; bi = idx; }
        }
        waveArgmax(bv, bi);
        if (lane == 0) { sTopBv[wv][k] = bv; sTopBi[wv][k] = bi; }
#pragma unroll
        for (int j = 0; j < 16; ++j) if (bi == lane + 32 * j) lv[j] = NEGBIG;
      }
    }

    // product combination: 256 candidates, 8 per lane, candidate id c = lane+32r
    float pv[8];
#pragma unroll
    for (int r = 0; r < 8; ++r) {
      int c = lane + 32 * r;
      pv[r] = sTopAv[tl][c >> 4] + sTopBv[wv][c & 15];
    }
    float selS[TOPK]; int selE[TOPK];
#pragma unroll
    for (int k = 0; k < TOPK; ++k) {
      float bv = pv[0]; int bi = lane;
#pragma unroll
      for (int r = 1; r < 8; ++r) {
        float v = pv[r]; int idx = lane + 32 * r;
        if (v > bv) { bv = v; bi = idx; }
      }
      waveArgmax(bv, bi);
      selS[k] = bv;
      selE[k] = sTopAi[tl][bi >> 4] * SKEYS + sTopBi[wv][bi & 15];
#pragma unroll
      for (int r = 0; r < 8; ++r) if (bi == lane + 32 * r) pv[r] = NEGBIG;
    }

    // softmax over the 16 selected (selS[0] is the max)
    float wgt[TOPK]; float sum = 0.f;
#pragma unroll
    for (int k = 0; k < TOPK; ++k) { wgt[k] = __expf(selS[k] - selS[0]); sum += wgt[k]; }
    const float inv = 1.f / sum;

    // gather expert_v rows and combine; lane owns dims {lane, lane+32, +64, +96}
    float a0 = 0.f, a1 = 0.f, a2 = 0.f, a3 = 0.f;
#pragma unroll
    for (int k = 0; k < TOPK; ++k) {
      const float* vr = expert_v + (size_t)selE[k] * HD;
      float wk = wgt[k] * inv;
      a0 += wk * vr[lane];
      a1 += wk * vr[lane + 32];
      a2 += wk * vr[lane + 64];
      a3 += wk * vr[lane + 96];
    }

    // shared-trunk gate: sigmoid(u . h) from f32 h
    const float* hr = hf + (size_t)token * DMODEL + (size_t)head * HD;
    float pd = hr[lane]      * uvec[lane]
             + hr[lane + 32] * uvec[lane + 32]
             + hr[lane + 64] * uvec[lane + 64]
             + hr[lane + 96] * uvec[lane + 96];
#pragma unroll
    for (int off = 16; off > 0; off >>= 1) pd += __shfl_xor(pd, off, 32);
    const float act = 1.f / (1.f + __expf(-pd));

    __bf16* mr = merged + (size_t)token * DMODEL + (size_t)head * HD;
    mr[lane]      = (__bf16)(act * a0);
    mr[lane + 32] = (__bf16)(act * a1);
    mr[lane + 64] = (__bf16)(act * a2);
    mr[lane + 96] = (__bf16)(act * a3);
  }
}

// ---------------------------------------------------------------------------
// Kernel 5: LayerNorm over 1024-wide rows (block of 256 = 8 waves per row)
// ---------------------------------------------------------------------------
__global__ __launch_bounds__(256) void ln_kernel(
    const float* __restrict__ P, const float* __restrict__ lw,
    const float* __restrict__ lb, float* __restrict__ out)
{
  __shared__ float sred[8];
  const int tid = threadIdx.x, lane = tid & 31, wv = tid >> 5;
  const float* p = P + (size_t)blockIdx.x * DMODEL;

  float x0 = p[tid], x1 = p[tid + 256], x2 = p[tid + 512], x3 = p[tid + 768];
  float s = x0 + x1 + x2 + x3;
#pragma unroll
  for (int off = 16; off > 0; off >>= 1) s += __shfl_xor(s, off, 32);
  if (lane == 0) sred[wv] = s;
  __syncthreads();
  if (tid == 0) { float t = 0; for (int i = 0; i < 8; ++i) t += sred[i]; sred[0] = t; }
  __syncthreads();
  const float mu = sred[0] * (1.f / DMODEL);
  __syncthreads();

  float d0 = x0 - mu, d1 = x1 - mu, d2 = x2 - mu, d3 = x3 - mu;
  float vs = d0 * d0 + d1 * d1 + d2 * d2 + d3 * d3;
#pragma unroll
  for (int off = 16; off > 0; off >>= 1) vs += __shfl_xor(vs, off, 32);
  if (lane == 0) sred[wv] = vs;
  __syncthreads();
  if (tid == 0) { float t = 0; for (int i = 0; i < 8; ++i) t += sred[i]; sred[0] = t; }
  __syncthreads();
  const float invs = rsqrtf(sred[0] * (1.f / DMODEL) + 1e-5f);

  float* o = out + (size_t)blockIdx.x * DMODEL;
  o[tid]       = d0 * invs * lw[tid]       + lb[tid];
  o[tid + 256] = d1 * invs * lw[tid + 256] + lb[tid + 256];
  o[tid + 512] = d2 * invs * lw[tid + 512] + lb[tid + 512];
  o[tid + 768] = d3 * invs * lw[tid + 768] + lb[tid + 768];
}

// ---------------------------------------------------------------------------
// launch
// ---------------------------------------------------------------------------
extern "C" void kernel_launch(void* const* d_in, const int* in_sizes, int n_in,
                              void* d_out, int out_size, void* d_ws, size_t ws_size,
                              hipStream_t stream) {
  (void)in_sizes; (void)n_in; (void)out_size; (void)ws_size;
  const float* x    = (const float*)d_in[0];
  const float* Win  = (const float*)d_in[1];
  const float* ka   = (const float*)d_in[2];
  const float* kb   = (const float*)d_in[3];
  const float* ush  = (const float*)d_in[4];
  const float* ev   = (const float*)d_in[5];
  const float* Wout = (const float*)d_in[6];
  const float* lnw  = (const float*)d_in[7];
  const float* lnb  = (const float*)d_in[8];
  float* out = (float*)d_out;

  char* ws = (char*)d_ws;
  size_t off = 0;
  auto take = [&](size_t bytes) -> char* {
    char* p = ws + off;
    off += (bytes + 255) & ~(size_t)255;
    return p;
  };

  const size_t nXW  = (size_t)NTOK * DMODEL;            // 8M elems
  const size_t nW   = (size_t)DMODEL * DMODEL;          // 1M elems
  const size_t nKey = (size_t)NHEAD * SKEYS * HD;       // 512K elems

  __bf16* x_bf    = (__bf16*)take(nXW * 2);
  __bf16* win_bf  = (__bf16*)take(nW * 2);
  __bf16* wout_bf = (__bf16*)take(nW * 2);
  __bf16* ka_bf   = (__bf16*)take(nKey * 2);
  __bf16* kb_bf   = (__bf16*)take(nKey * 2);
  float*  h_f     = (float*) take(nXW * 4);
  __bf16* h_bf    = (__bf16*)take(nXW * 2);
  __bf16* mrg_bf  = (__bf16*)take(nXW * 2);
  float*  proj    = (float*) take(nXW * 4);

  const int TB = 256;
  cvt_f32_bf16<<<(int)((nXW  + TB - 1) / TB), TB, 0, stream>>>(x,    x_bf,    (int)nXW);
  cvt_f32_bf16<<<(int)((nW   + TB - 1) / TB), TB, 0, stream>>>(Win,  win_bf,  (int)nW);
  cvt_f32_bf16<<<(int)((nW   + TB - 1) / TB), TB, 0, stream>>>(Wout, wout_bf, (int)nW);
  cvt_f32_bf16<<<(int)((nKey + TB - 1) / TB), TB, 0, stream>>>(ka,   ka_bf,   (int)nKey);
  cvt_f32_bf16<<<(int)((nKey + TB - 1) / TB), TB, 0, stream>>>(kb,   kb_bf,   (int)nKey);

  // h = x @ W_in^T   (f32 + bf16 outputs)
  gemm_nt_bf16<<<dim3(NTOK / 16, DMODEL / 512), 256, 0, stream>>>(
      x_bf, win_bf, h_f, h_bf);

  // fused PEER selection + expert combine -> merged (bf16)
  peer_select<<<dim3(NTOK / 16, NHEAD), 256, 0, stream>>>(
      h_bf, h_f, ka_bf, kb_bf, ush, ev, mrg_bf);

  // proj = merged @ W_out^T (f32)
  gemm_nt_bf16<<<dim3(NTOK / 16, DMODEL / 512), 256, 0, stream>>>(
      mrg_bf, wout_bf, proj, nullptr);

  // LayerNorm -> out
  ln_kernel<<<NTOK, 256, 0, stream>>>(proj, lnw, lnb, out);
}
